// Attention_37812892074210
// MI455X (gfx1250) — matrix operand order repaired and checked by reading, not compiled, not run on hardware
//
#include <hip/hip_runtime.h>

// ---------------------------------------------------------------------------
// Attention: O = softmax(Q K^T) K, fp32 in/out, bf16 WMMA compute (f32 accum)
// d_in[0] = encoder_outputs (K/V) [B, Tk, H]
// d_in[1] = decoder_hidden  (Q)   [B, Tq, H]
// d_out   = [ O: B*Tq*H floats | P: B*Tq*Tk floats ]
//
// Pipeline: (1) S = Q K^T (bf16 WMMA, f32 accum, double-buffered LDS)
//           (2) row softmax in place
//           (3) O = P K   (bf16 WMMA, K transposed into LDS)
// ---------------------------------------------------------------------------

typedef __attribute__((ext_vector_type(2)))  __bf16 bf16x2;
typedef __attribute__((ext_vector_type(8)))  __bf16 v8bf;
typedef __attribute__((ext_vector_type(16))) __bf16 v16bf;
typedef __attribute__((ext_vector_type(8)))  float  v8f;

#define NB   16
#define TQ   2048
#define TK   2048
#define HD   1024

#define BM   128      // output tile rows (q)
#define BN   128      // output tile cols (k or h)
#define BK   32       // reduction step (matches wmma K=32)
#define LP   40       // LDS row pitch in ushorts (80 B, multiple of 16 B)

// hardware bf16 convert (v_cvt_pk_bf16_f32 class), pack 4 floats -> 2 dwords
__device__ __forceinline__ uint2 cvt4(float4 v) {
  bf16x2 lo = { (__bf16)v.x, (__bf16)v.y };
  bf16x2 hi = { (__bf16)v.z, (__bf16)v.w };
  uint2 r;
  r.x = __builtin_bit_cast(unsigned int, lo);
  r.y = __builtin_bit_cast(unsigned int, hi);
  return r;
}

__device__ __forceinline__ unsigned short bfbits(float a) {
  __bf16 b = (__bf16)a;
  return __builtin_bit_cast(unsigned short, b);
}

__device__ __forceinline__ v16bf make16(v8bf lo, v8bf hi) {
  return __builtin_shufflevector(lo, hi, 0, 1, 2, 3, 4, 5, 6, 7,
                                         8, 9, 10, 11, 12, 13, 14, 15);
}

// ---------------------------------------------------------------------------
// Phase 1: S[b,q,k] = sum_h Q[b,q,h] * K[b,k,h]
// grid: (TK/BN, TQ/BM, NB), block: 256 (8 waves, 4x2 wave grid of 16x16 tiles)
// ---------------------------------------------------------------------------
__global__ __launch_bounds__(256) void qk_kernel(const float* __restrict__ Q,
                                                 const float* __restrict__ K,
                                                 float* __restrict__ S) {
  __shared__ unsigned short As[2][BM * LP];  // Q tile  [q][h] bf16
  __shared__ unsigned short Bs[2][BN * LP];  // K tile  [k][h] bf16

  const int tid  = threadIdx.x;
  const int lane = tid & 31;
  const int w    = tid >> 5;
  const int wq   = w >> 2;            // 0..1  (64 q-rows each)
  const int wk   = w & 3;             // 0..3  (32 k-cols each)
  const int m    = lane & 15;
  const int half = lane >> 4;

  const int lrow = tid >> 3;          // 0..31 (row within a 32-row pass)
  const int lc4  = tid & 7;           // 0..7  (float4 slot within 32-elem row)

  const int kBase = blockIdx.x * BN;
  const int qBase = blockIdx.y * BM;
  const int b     = blockIdx.z;

  const float* Qb = Q + (size_t)(b * TQ + qBase) * HD;
  const float* Kb = K + (size_t)(b * TK + kBase) * HD;

  v8f acc[4][2];
#pragma unroll
  for (int i = 0; i < 4; ++i)
#pragma unroll
    for (int jj = 0; jj < 2; ++jj) acc[i][jj] = (v8f){};

  const int NIT = HD / BK;            // 32
  float4 qreg[4], kreg[4];

  // ---- prologue: tile 0 -> LDS buffer 0
#pragma unroll
  for (int p = 0; p < 4; ++p) {
    const int row = p * 32 + lrow;
    qreg[p] = *(const float4*)(Qb + (size_t)row * HD + lc4 * 4);
    kreg[p] = *(const float4*)(Kb + (size_t)row * HD + lc4 * 4);
  }
#pragma unroll
  for (int p = 0; p < 4; ++p) {
    const int row = p * 32 + lrow;
    *(uint2*)&As[0][row * LP + lc4 * 4] = cvt4(qreg[p]);
    *(uint2*)&Bs[0][row * LP + lc4 * 4] = cvt4(kreg[p]);
  }
  __syncthreads();

  for (int it = 0; it < NIT; ++it) {
    const int cur = it & 1;
    const int nxt = cur ^ 1;

    // ---- issue next tile's global loads (overlap with WMMA below)
    if (it + 1 < NIT) {
      const int h0 = (it + 1) * BK;
#pragma unroll
      for (int p = 0; p < 4; ++p) {
        const int row = p * 32 + lrow;
        qreg[p] = *(const float4*)(Qb + (size_t)row * HD + h0 + lc4 * 4);
        kreg[p] = *(const float4*)(Kb + (size_t)row * HD + h0 + lc4 * 4);
      }
    }
    // ---- prefetch two tiles ahead into L2 (one lane per row)
    if (it + 2 < NIT && lc4 == 0) {
      const int h2 = (it + 2) * BK;
#pragma unroll
      for (int p = 0; p < 4; ++p) {
        const int row = p * 32 + lrow;
        __builtin_prefetch(Qb + (size_t)row * HD + h2, 0, 1);
        __builtin_prefetch(Kb + (size_t)row * HD + h2, 0, 1);
      }
    }

    // ---- compute on current buffer
    v16bf aF[4];
#pragma unroll
    for (int i = 0; i < 4; ++i) {
      const unsigned short* ap = &As[cur][(wq * 64 + i * 16 + m) * LP];
      aF[i] = make16(*(const v8bf*)(ap + half * 16),
                     *(const v8bf*)(ap + half * 16 + 8));
    }
    v16bf bF[2];
#pragma unroll
    for (int jj = 0; jj < 2; ++jj) {
      const unsigned short* bp = &Bs[cur][(wk * 32 + jj * 16 + m) * LP + half * 16];
      bF[jj] = make16(*(const v8bf*)(bp), *(const v8bf*)(bp + 8));
    }
#pragma unroll
    for (int i = 0; i < 4; ++i)
#pragma unroll
      for (int jj = 0; jj < 2; ++jj)
        acc[i][jj] = __builtin_amdgcn_wmma_f32_16x16x32_bf16(
            false, aF[i], false, bF[jj], (short)0, acc[i][jj], false, false);

    // ---- stash next tile into the other buffer
    if (it + 1 < NIT) {
#pragma unroll
      for (int p = 0; p < 4; ++p) {
        const int row = p * 32 + lrow;
        *(uint2*)&As[nxt][row * LP + lc4 * 4] = cvt4(qreg[p]);
        *(uint2*)&Bs[nxt][row * LP + lc4 * 4] = cvt4(kreg[p]);
      }
    }
    __syncthreads();
  }

#pragma unroll
  for (int i = 0; i < 4; ++i)
#pragma unroll
    for (int jj = 0; jj < 2; ++jj) {
      const int q0 = qBase + wq * 64 + i * 16;
      const int k0 = kBase + wk * 32 + jj * 16;
      float* out = S + (size_t)(b * TQ + q0 + half * 8) * TK + k0 + m;
#pragma unroll
      for (int j = 0; j < 8; ++j) out[(size_t)j * TK] = acc[i][jj][j];
    }
}

// ---------------------------------------------------------------------------
// Phase 2: in-place row softmax over last dim (TK) of S
// grid: NB*TQ, block: 256 (each thread owns 8 contiguous elements)
// ---------------------------------------------------------------------------
__global__ __launch_bounds__(256) void softmax_kernel(float* __restrict__ S) {
  __shared__ float red[256];
  const int t = threadIdx.x;
  float* row = S + (size_t)blockIdx.x * TK;

  float4 a = *(const float4*)(row + t * 8);
  float4 bq = *(const float4*)(row + t * 8 + 4);

  float mx = fmaxf(fmaxf(fmaxf(a.x, a.y), fmaxf(a.z, a.w)),
                   fmaxf(fmaxf(bq.x, bq.y), fmaxf(bq.z, bq.w)));
  red[t] = mx;
  __syncthreads();
  for (int s = 128; s > 0; s >>= 1) {
    if (t < s) red[t] = fmaxf(red[t], red[t + s]);
    __syncthreads();
  }
  mx = red[0];
  __syncthreads();

  a.x = __expf(a.x - mx);   a.y = __expf(a.y - mx);
  a.z = __expf(a.z - mx);   a.w = __expf(a.w - mx);
  bq.x = __expf(bq.x - mx); bq.y = __expf(bq.y - mx);
  bq.z = __expf(bq.z - mx); bq.w = __expf(bq.w - mx);

  red[t] = (a.x + a.y + a.z + a.w) + (bq.x + bq.y + bq.z + bq.w);
  __syncthreads();
  for (int s = 128; s > 0; s >>= 1) {
    if (t < s) red[t] += red[t + s];
    __syncthreads();
  }
  const float inv = 1.0f / red[0];

  a.x *= inv;  a.y *= inv;  a.z *= inv;  a.w *= inv;
  bq.x *= inv; bq.y *= inv; bq.z *= inv; bq.w *= inv;
  *(float4*)(row + t * 8) = a;
  *(float4*)(row + t * 8 + 4) = bq;
}

// ---------------------------------------------------------------------------
// Phase 3: O[b,q,h] = sum_k P[b,q,k] * K[b,k,h]
// grid: (HD/BN, TQ/BM, NB), block: 256
// K tile is transposed into LDS ([h][k]) so B fragments are 2x ds_load_b128
// ---------------------------------------------------------------------------
__global__ __launch_bounds__(256) void pv_kernel(const float* __restrict__ P,
                                                 const float* __restrict__ K,
                                                 float* __restrict__ O) {
  __shared__ unsigned short Ps[2][BM * LP];  // P tile  [q][k] bf16
  __shared__ unsigned short KT[2][BN * LP];  // K tile transposed [h][k] bf16

  const int tid  = threadIdx.x;
  const int lane = tid & 31;
  const int w    = tid >> 5;
  const int wq   = w >> 2;            // 0..1  (64 q-rows each)
  const int wk   = w & 3;             // 0..3  (32 h-cols each)
  const int m    = lane & 15;
  const int half = lane >> 4;

  const int lrow = tid >> 3;          // 0..31  (P tile row within pass)
  const int lc4  = tid & 7;           // 0..7   (P float4 slot)
  const int kRow = tid >> 5;          // 0..7   (K tile row within pass)
  const int kc4  = tid & 31;          // 0..31  (K float4 slot, 128 cols)

  const int hBase = blockIdx.x * BN;
  const int qBase = blockIdx.y * BM;
  const int b     = blockIdx.z;

  const float* Pb = P + (size_t)(b * TQ + qBase) * TK;
  const float* Kb = K + (size_t)b * TK * HD;

  v8f acc[4][2];
#pragma unroll
  for (int i = 0; i < 4; ++i)
#pragma unroll
    for (int jj = 0; jj < 2; ++jj) acc[i][jj] = (v8f){};

  const int NIT = TK / BK;            // 64
  float4 preg[4], kreg[4];

  // ---- prologue: tile 0
#pragma unroll
  for (int p = 0; p < 4; ++p) {
    preg[p] = *(const float4*)(Pb + (size_t)(p * 32 + lrow) * TK + lc4 * 4);
    kreg[p] = *(const float4*)(Kb + (size_t)(p * 8 + kRow) * HD + hBase + kc4 * 4);
  }
#pragma unroll
  for (int p = 0; p < 4; ++p) {
    *(uint2*)&Ps[0][(p * 32 + lrow) * LP + lc4 * 4] = cvt4(preg[p]);
    const int kr = p * 8 + kRow;
    const int hc = kc4 * 4;
    KT[0][(hc + 0) * LP + kr] = bfbits(kreg[p].x);
    KT[0][(hc + 1) * LP + kr] = bfbits(kreg[p].y);
    KT[0][(hc + 2) * LP + kr] = bfbits(kreg[p].z);
    KT[0][(hc + 3) * LP + kr] = bfbits(kreg[p].w);
  }
  __syncthreads();

  for (int kc = 0; kc < NIT; ++kc) {
    const int cur = kc & 1;
    const int nxt = cur ^ 1;

    // ---- issue next tile's global loads
    if (kc + 1 < NIT) {
      const int k0 = (kc + 1) * BK;
#pragma unroll
      for (int p = 0; p < 4; ++p) {
        preg[p] = *(const float4*)(Pb + (size_t)(p * 32 + lrow) * TK + k0 + lc4 * 4);
        kreg[p] = *(const float4*)(Kb + (size_t)(k0 + p * 8 + kRow) * HD + hBase + kc4 * 4);
      }
    }
    // ---- prefetch two tiles ahead
    if (kc + 2 < NIT && lc4 == 0) {
      const int k2 = (kc + 2) * BK;
#pragma unroll
      for (int p = 0; p < 4; ++p) {
        __builtin_prefetch(Pb + (size_t)(p * 32 + lrow) * TK + k2, 0, 1);
        __builtin_prefetch(Kb + (size_t)(k2 + p * 8 + kRow) * HD + hBase, 0, 1);
      }
    }

    // ---- compute on current buffer
    v16bf aF[4];
#pragma unroll
    for (int i = 0; i < 4; ++i) {
      const unsigned short* ap = &Ps[cur][(wq * 64 + i * 16 + m) * LP];
      aF[i] = make16(*(const v8bf*)(ap + half * 16),
                     *(const v8bf*)(ap + half * 16 + 8));
    }
    v16bf bF[2];
#pragma unroll
    for (int jj = 0; jj < 2; ++jj) {
      const unsigned short* bp = &KT[cur][(wk * 32 + jj * 16 + m) * LP + half * 16];
      bF[jj] = make16(*(const v8bf*)(bp), *(const v8bf*)(bp + 8));
    }
#pragma unroll
    for (int i = 0; i < 4; ++i)
#pragma unroll
      for (int jj = 0; jj < 2; ++jj)
        acc[i][jj] = __builtin_amdgcn_wmma_f32_16x16x32_bf16(
            false, aF[i], false, bF[jj], (short)0, acc[i][jj], false, false);

    // ---- stash next tile
    if (kc + 1 < NIT) {
#pragma unroll
      for (int p = 0; p < 4; ++p) {
        *(uint2*)&Ps[nxt][(p * 32 + lrow) * LP + lc4 * 4] = cvt4(preg[p]);
        const int kr = p * 8 + kRow;
        const int hc = kc4 * 4;
        KT[nxt][(hc + 0) * LP + kr] = bfbits(kreg[p].x);
        KT[nxt][(hc + 1) * LP + kr] = bfbits(kreg[p].y);
        KT[nxt][(hc + 2) * LP + kr] = bfbits(kreg[p].z);
        KT[nxt][(hc + 3) * LP + kr] = bfbits(kreg[p].w);
      }
    }
    __syncthreads();
  }

#pragma unroll
  for (int i = 0; i < 4; ++i)
#pragma unroll
    for (int jj = 0; jj < 2; ++jj) {
      const int q0 = qBase + wq * 64 + i * 16;
      const int h0 = hBase + wk * 32 + jj * 16;
      float* out = O + (size_t)(b * TQ + q0 + half * 8) * HD + h0 + m;
#pragma unroll
      for (int j = 0; j < 8; ++j) out[(size_t)j * HD] = acc[i][jj][j];
    }
}

// ---------------------------------------------------------------------------
extern "C" void kernel_launch(void* const* d_in, const int* in_sizes, int n_in,
                              void* d_out, int out_size, void* d_ws, size_t ws_size,
                              hipStream_t stream) {
  (void)in_sizes; (void)n_in; (void)out_size; (void)d_ws; (void)ws_size;

  const float* enc = (const float*)d_in[0];  // encoder_outputs [B, Tk, H]
  const float* dec = (const float*)d_in[1];  // decoder_hidden  [B, Tq, H]
  float* O = (float*)d_out;                              // [B, Tq, H]
  float* P = (float*)d_out + (size_t)NB * TQ * HD;       // [B, Tq, Tk]

  qk_kernel<<<dim3(TK / BN, TQ / BM, NB), 256, 0, stream>>>(dec, enc, P);
  softmax_kernel<<<NB * TQ, 256, 0, stream>>>(P);
  pv_kernel<<<dim3(HD / BN, TQ / BM, NB), 256, 0, stream>>>(P, enc, O);
}
